// SelfAttention_24266565222575
// MI455X (gfx1250) — matrix operand order repaired and checked by reading, not compiled, not run on hardware
//
#include <hip/hip_runtime.h>
#include <hip/hip_bf16.h>

// ---------------------------------------------------------------------------
// MI455X (gfx1250) sparse-attention pipeline, all matmuls via
// v_wmma_f32_16x16x32_bf16 (wave32, fp32 accumulate). GEMM K-loop is a
// ping-pong double-buffered pipeline (2x unroll) so loads land directly in
// the consuming register set: no rotation copies, partial s_wait_loadcnt.
// ---------------------------------------------------------------------------

typedef __attribute__((ext_vector_type(16))) __bf16 v16bf;
typedef __attribute__((ext_vector_type(8)))  __bf16 v8bf;
typedef __attribute__((ext_vector_type(8)))  float  v8f;

#define SEQ     2048
#define DIM     4096
#define NH      32
#define NKV     8
#define HD      128
#define NBLK    32
#define BLK     64
#define QKVN    6144          // 4096 q + 1024 k + 1024 v columns
#define NEGV    (-1.0e30f)
#define SCALE   0.08838834764831845f   // 1/sqrt(128)

// WMMA wrapper: D = A(16x32 bf16) * B(32x16 bf16) + C(16x16 f32)
__device__ __forceinline__ v8f wmma_bf16(v16bf a, v16bf b, v8f c) {
  return __builtin_amdgcn_wmma_f32_16x16x32_bf16(false, a, false, b,
                                                 (short)0, c, false, false);
}

// A-fragment: lane l<16 -> row l, K {k0..k0+15};
//             lane l>=16 -> row l-16, K {k0+16..k0+31}.
// `base` already includes row*stride + k0 + half*16; two 16B loads.
__device__ __forceinline__ v16bf load_a_frag(const __bf16* base) {
  v8bf lo = *(const v8bf*)(base);
  v8bf hi = *(const v8bf*)(base + 8);
  v16bf a;
#pragma unroll
  for (int i = 0; i < 8; ++i) { a[i] = lo[i]; a[i + 8] = hi[i]; }
  return a;
}

// ---------------------------------------------------------------------------
// Elementwise casts / transposes
// ---------------------------------------------------------------------------
__global__ void cast_f32_bf16(const float* __restrict__ src,
                              __bf16* __restrict__ dst, size_t n) {
  size_t i = blockIdx.x * (size_t)blockDim.x + threadIdx.x;
  if (i < n) dst[i] = (__bf16)src[i];
}

// dst[n*rows + k] = src[k*cols + n]   (weight [K,N] -> B^T [N,K] bf16)
__global__ void transpose_cast(const float* __restrict__ src,
                               __bf16* __restrict__ dst, int rows, int cols) {
  size_t i = blockIdx.x * (size_t)blockDim.x + threadIdx.x;
  size_t tot = (size_t)rows * cols;
  if (i >= tot) return;
  size_t n = i / rows, k = i % rows;
  dst[i] = (__bf16)src[k * (size_t)cols + n];
}

// ---------------------------------------------------------------------------
// GEMM: C[M,N] f32 = A[M,K] bf16 (row major) * BT[N,K] bf16
// Block = 256 threads = 8 waves; wave w covers rows [m0+16w, m0+16w+16) of a
// 128-row block tile and 64 columns (4 f32 accumulators). K-loop unrolled 2x
// with ping-pong fragment sets: stage X's loads issue while stage Y's WMMAs
// run, and vice versa; loads write directly into the set consumed next.
// ---------------------------------------------------------------------------
__global__ void __launch_bounds__(256)
gemm_bf16_tn(const __bf16* __restrict__ A, const __bf16* __restrict__ BT,
             float* __restrict__ C, int M, int N, int K) {
  const int lane = threadIdx.x & 31;
  const int wave = threadIdx.x >> 5;
  const int half = lane >> 4;
  const int ml   = lane & 15;
  const int m0   = blockIdx.x * 128 + wave * 16;
  const int n0   = blockIdx.y * 64;
  const __bf16* arow = A + (size_t)(m0 + ml) * K;
  const __bf16* brow[4];
#pragma unroll
  for (int sub = 0; sub < 4; ++sub)
    brow[sub] = BT + (size_t)(n0 + sub * 16 + ml) * K + half * 16;

  v8f acc[4] = {};

  // prologue: stage-0 fragments for k = 0
  v16bf a0 = load_a_frag(arow + half * 16);
  v16bf b0[4];
#pragma unroll
  for (int sub = 0; sub < 4; ++sub) b0[sub] = *(const v16bf*)(brow[sub]);

  for (int k0 = 0; k0 < K; k0 += 64) {
    const int k1 = k0 + 32;                          // always < K (K % 64 == 0)
    const int k2 = (k0 + 64 < K) ? (k0 + 64) : 0;    // wrap: dead load, no OOB
    __builtin_prefetch(arow + k0 + 1024, 0, 3);

    // stage 1 loads, stage 0 compute
    v16bf a1 = load_a_frag(arow + k1 + half * 16);
    v16bf b1[4];
#pragma unroll
    for (int sub = 0; sub < 4; ++sub) b1[sub] = *(const v16bf*)(brow[sub] + k1);
#pragma unroll
    for (int sub = 0; sub < 4; ++sub) acc[sub] = wmma_bf16(a0, b0[sub], acc[sub]);

    // stage 0 reloads (next iter), stage 1 compute
    a0 = load_a_frag(arow + k2 + half * 16);
#pragma unroll
    for (int sub = 0; sub < 4; ++sub) b0[sub] = *(const v16bf*)(brow[sub] + k2);
#pragma unroll
    for (int sub = 0; sub < 4; ++sub) acc[sub] = wmma_bf16(a1, b1[sub], acc[sub]);
  }

#pragma unroll
  for (int sub = 0; sub < 4; ++sub)
#pragma unroll
    for (int r = 0; r < 8; ++r)
      C[(size_t)(m0 + r + 8 * half) * N + n0 + sub * 16 + ml] = acc[sub][r];
}

// ---------------------------------------------------------------------------
// RoPE on q & k (fp32 in place in the fused QKV buffer, stride 6144),
// also emit bf16 copies: qb[32][2048][128], kb[8][2048][128].
// ---------------------------------------------------------------------------
__global__ void rope_qk(float* __restrict__ qkv,
                        const float* __restrict__ cosb,
                        const float* __restrict__ sinb,
                        __bf16* __restrict__ qb, __bf16* __restrict__ kb) {
  const size_t QP = (size_t)SEQ * NH  * (HD / 2);
  const size_t KP = (size_t)SEQ * NKV * (HD / 2);
  size_t i = blockIdx.x * (size_t)blockDim.x + threadIdx.x;
  if (i >= QP + KP) return;
  bool isq = i < QP;
  size_t j = isq ? i : i - QP;
  int nh   = isq ? NH : NKV;
  int pair = (int)(j % (HD / 2));
  int h    = (int)((j / (HD / 2)) % nh);
  int pos  = (int)(j / ((size_t)(HD / 2) * nh));
  int col0 = isq ? 0 : DIM;
  float* p = qkv + (size_t)pos * QKVN + col0 + h * HD + pair * 2;
  float x0 = p[0], x1 = p[1];
  float c = cosb[pos * (HD / 2) + pair];
  float s = sinb[pos * (HD / 2) + pair];
  float o0 = x0 * c - x1 * s;
  float o1 = x0 * s + x1 * c;
  p[0] = o0; p[1] = o1;
  __bf16* d = isq ? (qb + ((size_t)h * SEQ + pos) * HD + pair * 2)
                  : (kb + ((size_t)h * SEQ + pos) * HD + pair * 2);
  d[0] = (__bf16)o0; d[1] = (__bf16)o1;
}

// V -> bf16, transposed to [8][128][2048] so P·V B-fragments are contiguous.
__global__ void pack_v(const float* __restrict__ qkv, __bf16* __restrict__ vt) {
  size_t i = blockIdx.x * (size_t)blockDim.x + threadIdx.x;
  if (i >= (size_t)SEQ * NKV * HD) return;
  int d   = (int)(i % HD);
  int h   = (int)((i / HD) % NKV);
  int pos = (int)(i / ((size_t)HD * NKV));
  float v = qkv[(size_t)pos * QKVN + DIM + 1024 + h * HD + d];
  vt[((size_t)h * HD + d) * SEQ + pos] = (__bf16)v;
}

// Block means of roped K: kblk[8][32][128] fp32.
__global__ void kblk_mean(const float* __restrict__ qkv, float* __restrict__ kblk) {
  int i = blockIdx.x * blockDim.x + threadIdx.x;
  if (i >= NKV * NBLK * HD) return;
  int d = i & (HD - 1);
  int b = (i >> 7) & (NBLK - 1);
  int h = i >> 12;
  float s = 0.f;
#pragma unroll 4
  for (int t = 0; t < BLK; ++t)
    s += qkv[(size_t)(b * BLK + t) * QKVN + DIM + h * HD + d];
  kblk[i] = s * (1.0f / BLK);
}

// ---------------------------------------------------------------------------
// Top-8 block selection: one wave per (head, query). Lane b holds block b's
// score; 8 rounds of wave-max + first-set-lane pick -> 32-bit block mask.
// ---------------------------------------------------------------------------
__global__ void __launch_bounds__(128)
topk_blocks(const float* __restrict__ qkv, const float* __restrict__ kblk,
            unsigned* __restrict__ mask) {
  int gw   = blockIdx.x * (blockDim.x >> 5) + (threadIdx.x >> 5);
  int lane = threadIdx.x & 31;
  int h = gw >> 11;            // 2048 queries per head
  int q = gw & (SEQ - 1);
  if (h >= NH) return;
  const float* qp = qkv + (size_t)q * QKVN + h * HD;
  const float* kp = kblk + ((size_t)(h >> 2) * NBLK + lane) * HD;
  float s = 0.f;
#pragma unroll 4
  for (int d = 0; d < HD; ++d) s += qp[d] * kp[d];
  s *= SCALE;
  if (lane * BLK > q) s = NEGV;          // block-causal mask
  unsigned sel = 0;
#pragma unroll
  for (int it = 0; it < 8; ++it) {
    float m = s;
    for (int off = 16; off; off >>= 1) m = fmaxf(m, __shfl_xor(m, off));
    unsigned long long bal = __ballot(s == m);
    int w = __ffsll(bal) - 1;            // lowest index on ties, like lax.top_k
    sel |= 1u << w;
    if (lane == w) s = -__builtin_inff();
  }
  if (lane == 0) mask[(size_t)h * SEQ + q] = sel;
}

// ---------------------------------------------------------------------------
// Flash attention over selected blocks. One wave per (head, 16-query tile).
// S = Q·K^T via WMMA; online softmax with half-wave shuffles; P re-striped
// C-layout -> A-layout through LDS (same-wave DS ops are in-order; enforce
// with s_wait_dscnt); O += P·V via WMMA. Output bf16 [2048][4096].
// ---------------------------------------------------------------------------
__global__ void __launch_bounds__(128)
flash_attn(const __bf16* __restrict__ qb,    // [32][2048][128]
           const __bf16* __restrict__ kb,    // [8][2048][128]
           const __bf16* __restrict__ vt,    // [8][128][2048]
           const unsigned* __restrict__ mask,// [32][2048]
           __bf16* __restrict__ outb) {      // [2048][4096]
  __shared__ __bf16 plds[4][16][72];         // per-wave P tile, padded rows
  const int lane = threadIdx.x & 31;
  const int wave = threadIdx.x >> 5;
  const int half = lane >> 4;
  const int ml   = lane & 15;
  const int h    = blockIdx.y;
  const int hkv  = h >> 2;
  const int q0   = (blockIdx.x * 4 + wave) * 16;

  // Resident Q fragments (16 x 128 split into 4 K-chunks of 32)
  v16bf qa[4];
  const __bf16* qrow = qb + ((size_t)h * SEQ + q0 + ml) * HD;
#pragma unroll
  for (int kc = 0; kc < 4; ++kc) qa[kc] = load_a_frag(qrow + kc * 32 + half * 16);

  unsigned rm[8], um = 0;
#pragma unroll
  for (int r = 0; r < 8; ++r) {
    rm[r] = mask[(size_t)h * SEQ + q0 + r + 8 * half];
    um |= rm[r];
  }
  um |= __shfl_xor(um, 16);                  // union over all 16 rows

  float m_i[8], l_i[8];
#pragma unroll
  for (int r = 0; r < 8; ++r) { m_i[r] = NEGV; l_i[r] = 0.f; }
  v8f oacc[8] = {};

  const int bmax = (q0 + 15) >> 6;
  for (int b = 0; b <= bmax; ++b) {
    if (!((um >> b) & 1)) continue;

    // ---- S = Q · K^T  (16 q x 64 keys) ----
    v8f sacc[4] = {};
#pragma unroll
    for (int kc = 0; kc < 4; ++kc)
#pragma unroll
      for (int sub = 0; sub < 4; ++sub) {
        const __bf16* bp = kb + ((size_t)hkv * SEQ + b * BLK + sub * 16 + ml) * HD
                              + kc * 32 + half * 16;
        sacc[sub] = wmma_bf16(qa[kc], *(const v16bf*)bp, sacc[sub]);
      }

    // ---- mask + scale + online softmax ----
    float p[4][8], rmax[8];
#pragma unroll
    for (int r = 0; r < 8; ++r) {
      int  qpos = q0 + r + 8 * half;
      bool selr = (rm[r] >> b) & 1;
      float mx = NEGV;
#pragma unroll
      for (int sub = 0; sub < 4; ++sub) {
        int key = b * BLK + sub * 16 + ml;
        float s = (selr && key <= qpos) ? sacc[sub][r] * SCALE : NEGV;
        p[sub][r] = s;
        mx = fmaxf(mx, s);
      }
      mx = fmaxf(mx, __shfl_xor(mx, 1));
      mx = fmaxf(mx, __shfl_xor(mx, 2));
      mx = fmaxf(mx, __shfl_xor(mx, 4));
      mx = fmaxf(mx, __shfl_xor(mx, 8));     // row max within 16-lane half
      rmax[r] = mx;
    }
#pragma unroll
    for (int r = 0; r < 8; ++r) {
      float mnew = fmaxf(m_i[r], rmax[r]);
      float corr = __expf(m_i[r] - mnew);
      float rsum = 0.f;
#pragma unroll
      for (int sub = 0; sub < 4; ++sub) {
        float e = (p[sub][r] > 0.5f * NEGV) ? __expf(p[sub][r] - mnew) : 0.f;
        p[sub][r] = e;
        rsum += e;
      }
      rsum += __shfl_xor(rsum, 1);
      rsum += __shfl_xor(rsum, 2);
      rsum += __shfl_xor(rsum, 4);
      rsum += __shfl_xor(rsum, 8);
      l_i[r] = l_i[r] * corr + rsum;
      m_i[r] = mnew;
#pragma unroll
      for (int dt = 0; dt < 8; ++dt) oacc[dt][r] *= corr;
    }

    // ---- P: C-layout -> A-layout through LDS ----
#pragma unroll
    for (int sub = 0; sub < 4; ++sub)
#pragma unroll
      for (int r = 0; r < 8; ++r)
        plds[wave][r + 8 * half][sub * 16 + ml] = (__bf16)p[sub][r];
    asm volatile("s_wait_dscnt 0" ::: "memory");   // CDNA5 split DS counter

    v16bf pa[2];
#pragma unroll
    for (int kc2 = 0; kc2 < 2; ++kc2)
      pa[kc2] = load_a_frag(&plds[wave][ml][kc2 * 32 + half * 16]);

    // ---- O += P · V  (16 q x 128 d) ----
#pragma unroll
    for (int dt = 0; dt < 8; ++dt)
#pragma unroll
      for (int kc2 = 0; kc2 < 2; ++kc2) {
        const __bf16* vp = vt + ((size_t)hkv * HD + dt * 16 + ml) * SEQ
                              + b * BLK + kc2 * 32 + half * 16;
        oacc[dt] = wmma_bf16(pa[kc2], *(const v16bf*)vp, oacc[dt]);
      }
  }

  // ---- epilogue: normalize, store bf16 [seq][h*128+d] ----
#pragma unroll
  for (int r = 0; r < 8; ++r) {
    float inv = (l_i[r] > 0.f) ? 1.0f / l_i[r] : 0.f;
    int row = q0 + r + 8 * half;
#pragma unroll
    for (int dt = 0; dt < 8; ++dt)
      outb[(size_t)row * DIM + h * HD + dt * 16 + ml] = (__bf16)(oacc[dt][r] * inv);
  }
}

// ---------------------------------------------------------------------------
// Host-side launch
// ---------------------------------------------------------------------------
static inline dim3 grid1d(size_t n, int bs) { return dim3((unsigned)((n + bs - 1) / bs)); }

extern "C" void kernel_launch(void* const* d_in, const int* in_sizes, int n_in,
                              void* d_out, int out_size, void* d_ws, size_t ws_size,
                              hipStream_t stream) {
  (void)in_sizes; (void)n_in; (void)out_size; (void)ws_size;
  const float* x    = (const float*)d_in[0];
  const float* cosb = (const float*)d_in[1];
  const float* sinb = (const float*)d_in[2];
  const float* wq   = (const float*)d_in[3];
  const float* wk   = (const float*)d_in[4];
  const float* wv   = (const float*)d_in[5];
  const float* wo   = (const float*)d_in[6];
  float* out = (float*)d_out;

  char* ws = (char*)d_ws;
  size_t off = 0;
  auto alloc = [&](size_t bytes) -> void* {
    void* p = ws + off;
    off = (off + bytes + 255) & ~(size_t)255;
    return p;
  };
  __bf16*   xb    = (__bf16*)  alloc((size_t)SEQ * DIM * 2);        // 16 MB
  __bf16*   wqkvT = (__bf16*)  alloc((size_t)QKVN * DIM * 2);       // 48 MB
  __bf16*   woT   = (__bf16*)  alloc((size_t)DIM * DIM * 2);        // 32 MB
  float*    qkvf  = (float*)   alloc((size_t)SEQ * QKVN * 4);       // 48 MB
  __bf16*   qbf   = (__bf16*)  alloc((size_t)NH  * SEQ * HD * 2);   // 16 MB
  __bf16*   kbf   = (__bf16*)  alloc((size_t)NKV * SEQ * HD * 2);   //  4 MB
  __bf16*   vtb   = (__bf16*)  alloc((size_t)NKV * HD * SEQ * 2);   //  4 MB
  float*    kblk  = (float*)   alloc((size_t)NKV * NBLK * HD * 4);
  unsigned* msk   = (unsigned*)alloc((size_t)NH * SEQ * 4);
  __bf16*   attnb = (__bf16*)  alloc((size_t)SEQ * DIM * 2);        // 16 MB

  // 1) casts / weight transposes (B^T bf16 layouts for WMMA B-fragments)
  cast_f32_bf16<<<grid1d((size_t)SEQ * DIM, 256), 256, 0, stream>>>(x, xb, (size_t)SEQ * DIM);
  transpose_cast<<<grid1d((size_t)DIM * DIM, 256), 256, 0, stream>>>(wq, wqkvT, DIM, DIM);
  transpose_cast<<<grid1d((size_t)DIM * 1024, 256), 256, 0, stream>>>(wk, wqkvT + (size_t)DIM * DIM, DIM, 1024);
  transpose_cast<<<grid1d((size_t)DIM * 1024, 256), 256, 0, stream>>>(wv, wqkvT + (size_t)(DIM + 1024) * DIM, DIM, 1024);
  transpose_cast<<<grid1d((size_t)DIM * DIM, 256), 256, 0, stream>>>(wo, woT, DIM, DIM);

  // 2) fused QKV projection: [2048,4096] x [4096,6144] -> fp32
  gemm_bf16_tn<<<dim3(SEQ / 128, QKVN / 64), 256, 0, stream>>>(xb, wqkvT, qkvf, SEQ, QKVN, DIM);

  // 3) RoPE + bf16 repacks
  rope_qk<<<grid1d((size_t)SEQ * (NH + NKV) * (HD / 2), 256), 256, 0, stream>>>(qkvf, cosb, sinb, qbf, kbf);
  pack_v<<<grid1d((size_t)SEQ * NKV * HD, 256), 256, 0, stream>>>(qkvf, vtb);

  // 4) block means + top-8 selection
  kblk_mean<<<grid1d((size_t)NKV * NBLK * HD, 256), 256, 0, stream>>>(qkvf, kblk);
  topk_blocks<<<grid1d((size_t)NH * SEQ * 32, 128), 128, 0, stream>>>(qkvf, kblk, msk);

  // 5) sparse flash attention (one wave per 16-query tile)
  flash_attn<<<dim3(SEQ / 64, NH), 128, 0, stream>>>(qbf, kbf, vtb, msk, attnb);

  // 6) output projection: [2048,4096] x [4096,4096] -> d_out fp32
  gemm_bf16_tn<<<dim3(SEQ / 128, DIM / 64), 256, 0, stream>>>(attnb, woT, out, SEQ, DIM, DIM);
}
